// GlobalSAModule_52879637348768
// MI455X (gfx1250) — compile-verified, matrix operand order checked by reading, not run on hardware
//
#include <hip/hip_runtime.h>
#include <hip/hip_bf16.h>

typedef __attribute__((ext_vector_type(16))) __bf16 v16bf;
typedef __attribute__((ext_vector_type(8)))  float  v8f;

// ---------- helpers ----------

__device__ __forceinline__ unsigned short f2bf(float f) {
    unsigned int u = __float_as_uint(f);
    u += 0x7FFFu + ((u >> 16) & 1u);   // round-to-nearest-even
    return (unsigned short)(u >> 16);
}

// Index (in 16-bit elements) of matrix element (m, k) inside an LDS/global buffer
// stored as row-tiles x K-slabs of 16x32 bf16 A-fragments (512 ushorts per frag,
// 16 contiguous ushorts per lane).  Matches the CDNA5 16-bit A-matrix layout:
// lane = (m&15) + 16*((k%32)/8 & 1), per-lane value j encodes v=j/2,h=j&1 with
// k%32 = 8*g + 16*(v>>2) + 2*(v&3) + h.
__device__ __forceinline__ int frag_idx(int m, int k, int nk) {
    int kk   = k & 31;
    int lane = (m & 15) | (((kk >> 3) & 1) << 4);
    int j    = ((((kk >> 4) << 2) | ((kk & 7) >> 1)) << 1) | (kk & 1);
    return (((m >> 4) * nk + (k >> 5)) << 9) + (lane << 4) + j;
}

// ---------- weight pre-conversion: f32 row-major [K x N] -> bf16 B-fragment order ----------
// Fragment order: [ct][ks], 512 bf16 per fragment, 16 per lane (mirror of A layout).
__global__ __launch_bounds__(256) void psa_wconv(const float* __restrict__ W,
                                                 unsigned short* __restrict__ dst,
                                                 int Kreal, int N, int nk, int total) {
    int e = blockIdx.x * 256 + threadIdx.x;
    if (e >= total) return;
    int f = e >> 9, r = e & 511;
    int lane = r >> 4, j = r & 15;
    int v = j >> 1, h = j & 1;
    int ct = f / nk, ks = f - ct * nk;
    int k = ks * 32 + ((lane >> 4) << 3) + ((v >> 2) << 4) + ((v & 3) << 1) + h;
    int n = (ct << 4) + (lane & 15);
    float val = (k < Kreal) ? W[(size_t)k * N + n] : 0.0f;
    dst[e] = f2bf(val);
}

// ---------- output init: x_out=0 (ReLU => 0 is a valid max identity), pos_out=0, batch_out=arange ----------
__global__ __launch_bounds__(256) void psa_init_out(float* __restrict__ out, int out_size) {
    int i = blockIdx.x * 256 + threadIdx.x;
    int zlim = out_size < 16432 ? out_size : 16432;   // 16*1024 + 16*3
    if (i < zlim) out[i] = 0.0f;
    if (i < 16) {
        if (out_size >= 16464)                        // batch_out stored as int64
            ((long long*)(out + 16432))[i] = (long long)i;
        else if (out_size >= 16448)                   // fallback: stored as 16 scalars
            out[16432 + i] = (float)i;
    }
}

// ---------- GEMM tile: src (LDS A-frags) x wf (global B-frags) -> dst (LDS A-frags), bias+ReLU ----------
__device__ __forceinline__ void gemm_block(const unsigned short* __restrict__ src, int nkSrc,
                                           const unsigned short* __restrict__ wf,
                                           const float* __restrict__ bias,
                                           int nct, int nkDst,
                                           unsigned short* __restrict__ dst,
                                           int wid, int lane) {
#pragma unroll 1
    for (int ct = wid; ct < nct; ct += 8) {
        // Memory clobber: forbid hoisting the (ct-invariant) LDS A-fragment loads
        // out of this loop -- without it the compiler spills ~1024 VGPRs of
        // preloaded fragments to scratch and feeds WMMA from scratch reloads.
        asm volatile("" ::: "memory");
        v8f acc[8];
        v8f z = {0.f, 0.f, 0.f, 0.f, 0.f, 0.f, 0.f, 0.f};
#pragma unroll
        for (int rt = 0; rt < 8; ++rt) acc[rt] = z;

#pragma unroll 2
        for (int ks = 0; ks < nkSrc; ++ks) {
            v16bf b = *(const v16bf*)(wf + (((size_t)(ct * nkSrc + ks)) << 9) + (lane << 4));
            const unsigned short* s0 = src + ((size_t)ks << 9) + (lane << 4);
#pragma unroll
            for (int rt = 0; rt < 8; ++rt) {
                v16bf a = *(const v16bf*)(s0 + (((size_t)rt * nkSrc) << 9));
                acc[rt] = __builtin_amdgcn_wmma_f32_16x16x32_bf16(
                    false, a, false, b, (short)0, acc[rt], false, false);
            }
        }

        int col = (ct << 4) + (lane & 15);
        float bv = bias[col];
        int g8 = (lane >> 4) << 3;
#pragma unroll
        for (int rt = 0; rt < 8; ++rt) {
#pragma unroll
            for (int k = 0; k < 8; ++k) {
                int row = (rt << 4) + k + g8;
                float v = acc[rt][k] + bv;
                v = v > 0.0f ? v : 0.0f;
                dst[frag_idx(row, col, nkDst)] = f2bf(v);
            }
        }
    }
}

// ---------- final GEMM + segment-max reduction ----------
__device__ __forceinline__ void gemm_final(const unsigned short* __restrict__ src, int nkSrc,
                                           const unsigned short* __restrict__ wf,
                                           const float* __restrict__ bias, int nct,
                                           const int* __restrict__ seg,
                                           float* __restrict__ out,
                                           int wid, int lane) {
#pragma unroll 1
    for (int ct = wid; ct < nct; ct += 8) {
        asm volatile("" ::: "memory");   // see gemm_block: prevent A-frag hoist/spill
        v8f acc[8];
        v8f z = {0.f, 0.f, 0.f, 0.f, 0.f, 0.f, 0.f, 0.f};
#pragma unroll
        for (int rt = 0; rt < 8; ++rt) acc[rt] = z;

#pragma unroll 2
        for (int ks = 0; ks < nkSrc; ++ks) {
            v16bf b = *(const v16bf*)(wf + (((size_t)(ct * nkSrc + ks)) << 9) + (lane << 4));
            const unsigned short* s0 = src + ((size_t)ks << 9) + (lane << 4);
#pragma unroll
            for (int rt = 0; rt < 8; ++rt) {
                v16bf a = *(const v16bf*)(s0 + (((size_t)rt * nkSrc) << 9));
                acc[rt] = __builtin_amdgcn_wmma_f32_16x16x32_bf16(
                    false, a, false, b, (short)0, acc[rt], false, false);
            }
        }

        int col = (ct << 4) + (lane & 15);
        float bv = bias[col];
        int g8 = (lane >> 4) << 3;
        int curSeg = -1;
        float runMax = 0.0f;
#pragma unroll
        for (int rt = 0; rt < 8; ++rt) {
#pragma unroll
            for (int k = 0; k < 8; ++k) {
                int row = (rt << 4) + k + g8;          // rows visited in ascending order
                float v = acc[rt][k] + bv;
                v = v > 0.0f ? v : 0.0f;
                int s = seg[row];
                if (s != curSeg) {
                    if (curSeg >= 0)
                        atomicMax((unsigned int*)(out + curSeg * 1024 + col),
                                  __float_as_uint(runMax));
                    curSeg = s;
                    runMax = v;
                } else {
                    runMax = fmaxf(runMax, v);
                }
            }
        }
        if (curSeg >= 0)
            atomicMax((unsigned int*)(out + curSeg * 1024 + col), __float_as_uint(runMax));
    }
}

// ---------- fused MLP + segment-max kernel ----------
// 1024 blocks x 256 threads (8 wave32), 128 rows per block.
// Dynamic LDS: seg(512B) | h1 frags (64KB) | {A frags (72KB) / h2 frags (128KB)} = 197120 B
__global__ __launch_bounds__(256, 1) void psa_fused(
    const float* __restrict__ x, const float* __restrict__ pos,
    const long long* __restrict__ batch,
    const unsigned short* __restrict__ w1f, const float* __restrict__ b1,
    const unsigned short* __restrict__ w2f, const float* __restrict__ b2,
    const unsigned short* __restrict__ w3f, const float* __restrict__ b3,
    float* __restrict__ out) {
    extern __shared__ char smem[];
    int* seg            = (int*)smem;
    unsigned short* h1  = (unsigned short*)(smem + 512);
    unsigned short* af  = (unsigned short*)(smem + 512 + 65536);
    unsigned short* h2  = af;   // reuse A region after layer 1

    const int tid = threadIdx.x;
    const int wid = tid >> 5;
    const int lane = tid & 31;
    const int rowBase = blockIdx.x * 128;

    if (tid < 128) seg[tid] = (int)batch[rowBase + tid];

    // Load x||pos tile, convert to bf16, scatter into A-fragment layout (K padded 259->288, 9 slabs)
    for (int e = tid; e < 128 * 288; e += 256) {
        int m = e / 288, k = e - m * 288;
        int gr = rowBase + m;
        float v = 0.0f;
        if (k < 256)      v = x[(size_t)gr * 256 + k];
        else if (k < 259) v = pos[(size_t)gr * 3 + (k - 256)];
        af[frag_idx(m, k, 9)] = f2bf(v);
    }
    __syncthreads();

    gemm_block(af, 9,  w1f, b1, 16, 8,  h1, wid, lane);   // [128,288]x[288,256] -> h1
    __syncthreads();
    gemm_block(h1, 8,  w2f, b2, 32, 16, h2, wid, lane);   // [128,256]x[256,512] -> h2 (over A)
    __syncthreads();
    gemm_final(h2, 16, w3f, b3, 64, seg, out, wid, lane); // [128,512]x[512,1024] -> segmax
}

// ---------- host launch ----------
extern "C" void kernel_launch(void* const* d_in, const int* in_sizes, int n_in,
                              void* d_out, int out_size, void* d_ws, size_t ws_size,
                              hipStream_t stream) {
    const float*     x     = (const float*)d_in[0];
    const float*     pos   = (const float*)d_in[1];
    const long long* batch = (const long long*)d_in[2];
    const float*     W1    = (const float*)d_in[3];
    const float*     b1    = (const float*)d_in[4];
    const float*     W2    = (const float*)d_in[5];
    const float*     b2    = (const float*)d_in[6];
    const float*     W3    = (const float*)d_in[7];
    const float*     b3    = (const float*)d_in[8];
    float* out = (float*)d_out;

    const int N = in_sizes[0] / 256;   // 131072

    unsigned short* ws  = (unsigned short*)d_ws;
    unsigned short* w1f = ws;                       // 288*256  = 73728  elems
    unsigned short* w2f = ws + 73728;               // 256*512  = 131072 elems
    unsigned short* w3f = ws + 73728 + 131072;      // 512*1024 = 524288 elems

    psa_wconv<<<(73728  + 255) / 256, 256, 0, stream>>>(W1, w1f, 259, 256,  9,  73728);
    psa_wconv<<<(131072 + 255) / 256, 256, 0, stream>>>(W2, w2f, 256, 512,  8,  131072);
    psa_wconv<<<(524288 + 255) / 256, 256, 0, stream>>>(W3, w3f, 512, 1024, 16, 524288);

    psa_init_out<<<(16464 + 255) / 256, 256, 0, stream>>>(out, out_size);

    const size_t smem = 512 + 65536 + 131072;   // 197120 B < 320 KB/WGP
    psa_fused<<<N / 128, 256, smem, stream>>>(x, pos, batch,
                                              w1f, b1, w2f, b2, w3f, b3, out);
}
